// SparseAttention_18622978195511
// MI455X (gfx1250) — compile-verified, matrix operand order checked by reading, not run on hardware
//
#include <hip/hip_runtime.h>
#include <hip/hip_bf16.h>

// ---------------------------------------------------------------------------
// SparseAttention (strided) == dense attention (permutation cancels):
//   out = softmax((Q K^T)/16) V,  batch=2, t=4096, d=1024, fp32 in/out.
// Compute-bound -> bf16 WMMA flash attention.
//  - pre-pass converts Q,K to bf16 and V to *transposed* bf16 [b][dim][key]
//  - all WMMA fragments are 2x ds_load_b128 (k-contiguous in LDS)
//  - K and V^T tiles software-pipelined through registers
// ---------------------------------------------------------------------------

typedef __attribute__((ext_vector_type(16))) __bf16 v16bf;
typedef __attribute__((ext_vector_type(8)))  float  v8f;
typedef __attribute__((ext_vector_type(4)))  unsigned v4u;

#define BATCH   2
#define T_CTX   4096
#define D_DIM   1024
#define BM      32      // query rows per workgroup
#define BN      64      // keys per iteration
#define DCHUNK  128     // contraction chunk for QK^T
#define SCALE   0.0625f // 1/sqrt(1024/4)
#define SMEM_BYTES 161280

__device__ __forceinline__ unsigned short f32_bf16(float f) {
  unsigned u = __float_as_uint(f);
  u += 0x7FFFu + ((u >> 16) & 1u);      // round-to-nearest-even
  return (unsigned short)(u >> 16);
}

// 16-bit WMMA fragment from a row-major LDS row (k-contiguous source).
// Elements 0..7 <-> k0..k0+7 ; elements 8..15 <-> k0+16..k0+23 (lane-half +8).
__device__ __forceinline__ v16bf frag_ld(const unsigned short* rowbase, int k0) {
  union { v4u u[2]; v16bf v; } t;
  t.u[0] = *(const v4u*)(rowbase + k0);
  t.u[1] = *(const v4u*)(rowbase + k0 + 16);
  return t.v;
}

__global__ void cvt_f32_bf16(const float* __restrict__ s,
                             unsigned short* __restrict__ d, int npairs) {
  int i = blockIdx.x * blockDim.x + threadIdx.x;
  if (i < npairs) {
    float2 f = ((const float2*)s)[i];
    unsigned lo = f32_bf16(f.x), hi = f32_bf16(f.y);
    ((unsigned*)d)[i] = lo | (hi << 16);
  }
}

// V [b][t][d] f32  ->  V^T [b][d][t] bf16 (one-time transpose pre-pass).
// Consecutive threads cover consecutive d -> coalesced reads; 16B writes.
__global__ void cvt_transpose_v(const float* __restrict__ s,
                                unsigned short* __restrict__ d) {
  size_t i = (size_t)blockIdx.x * blockDim.x + threadIdx.x;
  int dim  = (int)(i & (D_DIM - 1));
  size_t r = i >> 10;
  int tg   = (int)(r & (T_CTX / 8 - 1));       // group of 8 keys
  int b    = (int)(r >> 9);
  const float* sp = s + ((size_t)b * T_CTX + (size_t)tg * 8) * D_DIM + dim;
  union { unsigned short h[8]; v4u u; } o;
#pragma unroll
  for (int j = 0; j < 8; ++j) o.h[j] = f32_bf16(sp[(size_t)j * D_DIM]);
  *(v4u*)(d + ((size_t)b * D_DIM + dim) * T_CTX + (size_t)tg * 8) = o.u;
}

__global__ __launch_bounds__(256, 1)
void flash_attn_kernel(const unsigned short* __restrict__ Q,
                       const unsigned short* __restrict__ K,
                       const unsigned short* __restrict__ Vt,
                       float* __restrict__ O) {
  extern __shared__ char smem[];
  unsigned short* sQ    = (unsigned short*)(smem);            // 32x1024 bf16 (64K)
  unsigned short* sK    = (unsigned short*)(smem + 65536);    // 64x128 bf16 (16K)
  float*          sS    = (float*)         (smem + 81920);    // 32x64 f32 (8K)
  unsigned short* sP    = (unsigned short*)(smem + 90112);    // 32x64 bf16 (4K)
  unsigned short* sVt   = (unsigned short*)(smem + 94208);    // 1024x32 bf16 (64K)
  float*          sM    = (float*)(smem + 159744);            // 32 f32 each
  float*          sL    = (float*)(smem + 159872);
  float*          sNewM = (float*)(smem + 160000);
  float*          sAlpha= (float*)(smem + 160128);
  float*          sRed  = (float*)(smem + 160256);            // 32x8 f32

  const int tid  = threadIdx.x;
  const int wave = tid >> 5;
  const int lane = tid & 31;
  const int b    = blockIdx.y;
  const int q0   = blockIdx.x * BM;

  const unsigned short* Qg  = Q  + ((size_t)b * T_CTX + q0) * D_DIM;
  const unsigned short* Kg  = K  + (size_t)b * T_CTX * D_DIM;
  const unsigned short* Vtg = Vt + (size_t)b * D_DIM * T_CTX;

  // -------- stage Q block: 32x1024 bf16 = contiguous 64 KB ---------------
  {
    v4u tmp[16];
    const v4u* src = (const v4u*)Qg;
#pragma unroll
    for (int i = 0; i < 16; ++i) tmp[i] = src[tid + i * 256];
    v4u* dst = (v4u*)sQ;
#pragma unroll
    for (int i = 0; i < 16; ++i) dst[tid + i * 256] = tmp[i];
  }
  if (tid < BM) { sM[tid] = -INFINITY; sL[tid] = 0.0f; }

  const v8f vzero = {};
  v8f acc[16];                 // 32 rows x 128 dims of O per wave
#pragma unroll
  for (int i = 0; i < 16; ++i) acc[i] = vzero;

  const int mt    = wave >> 2;           // S tile row (0..1)
  const int nt    = wave & 3;            // S tile col (0..3)
  const int lmod  = lane & 15;
  const int kbase = (lane < 16) ? 0 : 8; // ISA 16-bit fragment K pattern
  const int rhalf = (lane < 16) ? 0 : 8; // C/D tile row half

  // preload first K chunk (64 keys x 128 dims) into registers
  v4u kreg[4];
#pragma unroll
  for (int i = 0; i < 4; ++i) {
    int idx = tid + i * 256;
    int r = idx >> 4, c = idx & 15;
    kreg[i] = *(const v4u*)(Kg + (size_t)r * D_DIM + c * 8);
  }

  // PV compute for one 32-key chunk from sP/sVt
  auto pv_compute = [&](int kt) {
    v16bf aP0 = frag_ld(sP + (0 * 16 + lmod) * BN + kt * 32, kbase);
    v16bf aP1 = frag_ld(sP + (1 * 16 + lmod) * BN + kt * 32, kbase);
#pragma unroll
    for (int n2 = 0; n2 < 8; ++n2) {
      v16bf bV = frag_ld(sVt + (wave * 128 + n2 * 16 + lmod) * 32, kbase);
      acc[n2] = __builtin_amdgcn_wmma_f32_16x16x32_bf16(
          false, aP0, false, bV, (short)0, acc[n2], false, false);
      acc[8 + n2] = __builtin_amdgcn_wmma_f32_16x16x32_bf16(
          false, aP1, false, bV, (short)0, acc[8 + n2], false, false);
    }
  };

  for (int n0 = 0; n0 < T_CTX; n0 += BN) {
    // ---------------- S = Q K^T (chunked over D, pipelined K) -------------
    v8f cS = vzero;
    for (int dk = 0; dk < D_DIM; dk += DCHUNK) {
      __syncthreads();                   // previous chunk's reads retired
#pragma unroll
      for (int i = 0; i < 4; ++i) ((v4u*)sK)[tid + i * 256] = kreg[i];
      {                                  // issue next chunk's global loads
        int n0n = n0, dkn = dk + DCHUNK;
        if (dkn == D_DIM) { dkn = 0; n0n = n0 + BN; }
        if (n0n < T_CTX) {
          const unsigned short* kb = Kg + (size_t)n0n * D_DIM + dkn;
#pragma unroll
          for (int i = 0; i < 4; ++i) {
            int idx = tid + i * 256;
            int r = idx >> 4, c = idx & 15;
            kreg[i] = *(const v4u*)(kb + (size_t)r * D_DIM + c * 8);
          }
        }
      }
      __syncthreads();                   // sK visible to all
#pragma unroll
      for (int ks = 0; ks < 4; ++ks) {
        const int ka = ks * 32;
        v16bf a  = frag_ld(sQ + (mt * 16 + lmod) * D_DIM + dk, ka + kbase);
        v16bf bb = frag_ld(sK + (nt * 16 + lmod) * DCHUNK, ka + kbase);
        cS = __builtin_amdgcn_wmma_f32_16x16x32_bf16(false, a, false, bb,
                                                     (short)0, cS, false, false);
      }
    }

    // prefetch V^T tile kt=0 (1024 dims x 32 keys) into registers
    v4u vtmp[16];
#pragma unroll
    for (int i = 0; i < 16; ++i) {
      int idx = tid + i * 256;
      int dd = idx >> 2, cg = idx & 3;
      vtmp[i] = *(const v4u*)(Vtg + (size_t)dd * T_CTX + n0 + cg * 8);
    }

    __syncthreads();
    {                                    // spill S tile (C layout -> LDS)
      int col = nt * 16 + lmod;
      int rb  = mt * 16 + rhalf;
#pragma unroll
      for (int g = 0; g < 8; ++g) sS[(rb + g) * BN + col] = cS[g];
    }
    __syncthreads();

    // ---------------- online softmax (8 threads per row) ----------------
    {
      const int row = tid >> 3;
      const int sub = tid & 7;
      float vals[8];
      float pm = -INFINITY;
#pragma unroll
      for (int j = 0; j < 8; ++j) {
        vals[j] = sS[row * BN + sub + j * 8] * SCALE;
        pm = fmaxf(pm, vals[j]);
      }
      sRed[row * 8 + sub] = pm;
      __syncthreads();
      if (sub == 0) {
        float mx = sRed[row * 8];
#pragma unroll
        for (int j = 1; j < 8; ++j) mx = fmaxf(mx, sRed[row * 8 + j]);
        float mprev = sM[row];
        float mnew  = fmaxf(mprev, mx);
        sAlpha[row] = __expf(mprev - mnew);
        sNewM[row]  = mnew;
        sM[row]     = mnew;
      }
      __syncthreads();
      float mnew = sNewM[row];
      float psum = 0.0f;
#pragma unroll
      for (int j = 0; j < 8; ++j) {
        float p = __expf(vals[j] - mnew);
        psum += p;
        sP[row * BN + sub + j * 8] = f32_bf16(p);
      }
      sRed[row * 8 + sub] = psum;
      __syncthreads();
      if (sub == 0) {
        float s = sRed[row * 8];
#pragma unroll
        for (int j = 1; j < 8; ++j) s += sRed[row * 8 + j];
        sL[row] = sL[row] * sAlpha[row] + s;
      }
    }
    __syncthreads();

    // ---------------- rescale O accumulators ----------------
    {
      float al[2][8];
#pragma unroll
      for (int m2 = 0; m2 < 2; ++m2)
#pragma unroll
        for (int g = 0; g < 8; ++g) al[m2][g] = sAlpha[m2 * 16 + rhalf + g];
#pragma unroll
      for (int t = 0; t < 16; ++t) {
        int m2 = t >> 3;
#pragma unroll
        for (int g = 0; g < 8; ++g) acc[t][g] *= al[m2][g];
      }
    }

    // ---------------- O += P V (pipelined V^T tiles) ----------------
    {
      v4u* dst = (v4u*)sVt;
#pragma unroll
      for (int i = 0; i < 16; ++i) dst[tid + i * 256] = vtmp[i];  // kt=0 tile
#pragma unroll
      for (int i = 0; i < 16; ++i) {     // prefetch kt=1 tile
        int idx = tid + i * 256;
        int dd = idx >> 2, cg = idx & 3;
        vtmp[i] = *(const v4u*)(Vtg + (size_t)dd * T_CTX + (n0 + 32) + cg * 8);
      }
      __syncthreads();
      pv_compute(0);
      __syncthreads();                   // kt=0 readers retired
#pragma unroll
      for (int i = 0; i < 16; ++i) dst[tid + i * 256] = vtmp[i];  // kt=1 tile
      __syncthreads();
      pv_compute(1);
    }
  }

  __syncthreads();
  // ---------------- epilogue: normalize by l and store fp32 ----------------
  {
    float inv[2][8];
#pragma unroll
    for (int m2 = 0; m2 < 2; ++m2)
#pragma unroll
      for (int g = 0; g < 8; ++g) inv[m2][g] = 1.0f / sL[m2 * 16 + rhalf + g];
#pragma unroll
    for (int t = 0; t < 16; ++t) {
      int m2 = t >> 3, n2 = t & 7;
      int dim = wave * 128 + n2 * 16 + lmod;
#pragma unroll
      for (int g = 0; g < 8; ++g) {
        int row = q0 + m2 * 16 + rhalf + g;
        O[((size_t)b * T_CTX + row) * D_DIM + dim] = acc[t][g] * inv[m2][g];
      }
    }
  }
}

extern "C" void kernel_launch(void* const* d_in, const int* in_sizes, int n_in,
                              void* d_out, int out_size, void* d_ws, size_t ws_size,
                              hipStream_t stream) {
  (void)in_sizes; (void)n_in; (void)out_size; (void)ws_size;
  const float* q = (const float*)d_in[0];
  const float* k = (const float*)d_in[1];
  const float* v = (const float*)d_in[2];

  const size_t elems = (size_t)BATCH * T_CTX * D_DIM;   // 8,388,608 per tensor
  unsigned short* wsQ  = (unsigned short*)d_ws;         // bf16 Q  [b][t][d]
  unsigned short* wsK  = wsQ + elems;                   // bf16 K  [b][t][d]
  unsigned short* wsVt = wsK + elems;                   // bf16 V^T [b][d][t]

  const int pairs = (int)(elems / 2);
  const int cvt_blocks = (pairs + 255) / 256;
  cvt_f32_bf16<<<cvt_blocks, 256, 0, stream>>>(q, wsQ, pairs);
  cvt_f32_bf16<<<cvt_blocks, 256, 0, stream>>>(k, wsK, pairs);
  const int trv_blocks = (int)((elems / 8 + 255) / 256);
  cvt_transpose_v<<<trv_blocks, 256, 0, stream>>>(v, wsVt);

  hipError_t attr_rc = hipFuncSetAttribute(
      (const void*)flash_attn_kernel,
      hipFuncAttributeMaxDynamicSharedMemorySize, SMEM_BYTES);
  (void)attr_rc;
  dim3 grid(T_CTX / BM, BATCH);
  flash_attn_kernel<<<grid, 256, SMEM_BYTES, stream>>>(wsQ, wsK, wsVt,
                                                       (float*)d_out);
}
